// ForgettingMonotonicConvBertSelfAttention_68616397521424
// MI455X (gfx1250) — compile-verified, matrix operand order checked by reading, not run on hardware
//
#include <hip/hip_runtime.h>
#include <math.h>

// ---- problem constants (from reference) ----
#define BB   4
#define SS   1024
#define HID  768
#define AH   384
#define NH   6
#define DD   64
#define KS   9
#define PADK 4

typedef __attribute__((ext_vector_type(16))) _Float16 v16h;
typedef __attribute__((ext_vector_type(8)))  float    v8f;
typedef __attribute__((ext_vector_type(4)))  int      v4i;

union Frag16 { v16h v; unsigned u[8]; };

// ---- CDNA5 async Global->LDS DMA (ASYNCcnt) with safe fallback ----
#if defined(__gfx1250__) && __has_builtin(__builtin_amdgcn_global_load_async_to_lds_b128)
#define USE_ASYNC 1
#else
#define USE_ASYNC 0
#endif

__device__ __forceinline__ void copy16_g2l(const _Float16* g, _Float16* l) {
#if USE_ASYNC
    __builtin_amdgcn_global_load_async_to_lds_b128(
        (__attribute__((address_space(1))) v4i*)(v4i*)g,
        (__attribute__((address_space(3))) v4i*)(v4i*)l, 0, 0);
#else
    *(uint4*)l = *(const uint4*)g;
#endif
}
__device__ __forceinline__ void wait_copies() {
#if USE_ASYNC
#if __has_builtin(__builtin_amdgcn_s_wait_asynccnt)
    __builtin_amdgcn_s_wait_asynccnt(0);
#else
    asm volatile("s_wait_asynccnt 0x0" ::: "memory");
#endif
#endif
}

// =====================================================================
// Uniform guard-free WMMA GEMM: C = alpha * A(MxK,f16) @ B(NxK,f16)^T
//   + bias; optional f32 (Cf) and f16 (Ch) outputs. Batched over
//   blockIdx.z = b*Hdim + h. M,N multiples of 64 (by grid), K mult of 32.
// 128 threads = 4 waves; 64x64 block tile; each wave 32x32 via 2x2
// v_wmma_f32_16x16x32_f16; double-buffered LDS fed by async b128 DMA.
// =====================================================================
#define TK   32
#define LSTR 40   // halves per LDS row (80B: 4x16B data + 16B pad)

__global__ __launch_bounds__(128) void gemm_wmma_f16(
    const _Float16* __restrict__ A, int lda,
    const _Float16* __restrict__ B, int ldb,
    float* __restrict__ Cf, _Float16* __restrict__ Ch, int ldc,
    const float* __restrict__ bias, int nbias,
    int K, float alpha, int Hdim,
    long sA0, long sA1, long sB0, long sB1, long sC0, long sC1)
{
    __shared__ __align__(16) _Float16 Ah[2][64 * LSTR];
    __shared__ __align__(16) _Float16 Bh[2][64 * LSTR];

    const int z  = blockIdx.z;
    const int bb = z / Hdim, hh = z % Hdim;
    A += (long)bb * sA0 + (long)hh * sA1;
    B += (long)bb * sB0 + (long)hh * sB1;
    if (Cf) Cf += (long)bb * sC0 + (long)hh * sC1;
    if (Ch) Ch += (long)bb * sC0 + (long)hh * sC1;

    const int tileM = blockIdx.y * 64;
    const int tileN = blockIdx.x * 64;
    const int tid  = threadIdx.x;
    const int wave = tid >> 5;
    const int lane = tid & 31;
    const int wm = (wave >> 1) * 32;
    const int wn = (wave & 1) * 32;
    const int lrow = lane & 15;
    const int kbase = (lane >> 4) * 8;   // ISA 16-bit fragment K grouping
    const int rowadd = (lane >> 4) * 8;  // C/D: M = r (+8 for hi lanes)

    // this thread's two 16B chunks per tile: chunk c -> row c>>2, quad c&3
    const int c0 = tid * 2;
    const int ar0 = (c0) >> 2,     aq0 = ((c0) & 3) * 8;
    const int ar1 = (c0 + 1) >> 2, aq1 = ((c0 + 1) & 3) * 8;
    const _Float16* Ag = A + (long)tileM * lda;
    const _Float16* Bg = B + (long)tileN * ldb;

    v8f acc[2][2] = {};
    const int nk = K / TK;

    // prologue: stage k-step 0 into buffer 0
    copy16_g2l(Ag + (long)ar0 * lda + aq0, &Ah[0][ar0 * LSTR + aq0]);
    copy16_g2l(Ag + (long)ar1 * lda + aq1, &Ah[0][ar1 * LSTR + aq1]);
    copy16_g2l(Bg + (long)ar0 * ldb + aq0, &Bh[0][ar0 * LSTR + aq0]);
    copy16_g2l(Bg + (long)ar1 * ldb + aq1, &Bh[0][ar1 * LSTR + aq1]);

    for (int kt = 0; kt < nk; ++kt) {
        wait_copies();
        __syncthreads();
        if (kt + 1 < nk) {               // prefetch next k-step, other buffer
            const int nb = (kt + 1) & 1;
            const int ko = (kt + 1) * TK;
            copy16_g2l(Ag + (long)ar0 * lda + ko + aq0, &Ah[nb][ar0 * LSTR + aq0]);
            copy16_g2l(Ag + (long)ar1 * lda + ko + aq1, &Ah[nb][ar1 * LSTR + aq1]);
            copy16_g2l(Bg + (long)ar0 * ldb + ko + aq0, &Bh[nb][ar0 * LSTR + aq0]);
            copy16_g2l(Bg + (long)ar1 * ldb + ko + aq1, &Bh[nb][ar1 * LSTR + aq1]);
        }
        const int cb = kt & 1;
        Frag16 a[2], b[2];
        #pragma unroll
        for (int t = 0; t < 2; ++t) {
            const _Float16* ap = &Ah[cb][(wm + t * 16 + lrow) * LSTR];
            const _Float16* bp = &Bh[cb][(wn + t * 16 + lrow) * LSTR];
            #pragma unroll
            for (int v = 0; v < 8; ++v) {
                int kp = kbase + ((v < 4) ? (v * 2) : (16 + (v - 4) * 2));
                a[t].u[v] = *(const unsigned*)(ap + kp);
                b[t].u[v] = *(const unsigned*)(bp + kp);
            }
        }
        #pragma unroll
        for (int tm = 0; tm < 2; ++tm)
            #pragma unroll
            for (int tn = 0; tn < 2; ++tn)
                acc[tm][tn] = __builtin_amdgcn_wmma_f32_16x16x32_f16(
                    false, a[tm].v, false, b[tn].v, (short)0, acc[tm][tn],
                    false, false);
        __syncthreads();                 // reads done before buffer reuse
    }

    #pragma unroll
    for (int tm = 0; tm < 2; ++tm)
        #pragma unroll
        for (int tn = 0; tn < 2; ++tn) {
            int col = tileN + wn + tn * 16 + lrow;
            #pragma unroll
            for (int r = 0; r < 8; ++r) {
                int row = tileM + wm + tm * 16 + r + rowadd;
                float v = acc[tm][tn][r] * alpha;
                if (bias && col < nbias) v += bias[col];
                if (Cf) Cf[(long)row * ldc + col] = v;
                if (Ch) Ch[(long)row * ldc + col] = (_Float16)v;
            }
        }
}

// ---- pre-pass kernels -------------------------------------------------
__global__ void cvt_h_k(const float* __restrict__ in, _Float16* __restrict__ out,
                        long n)
{
    long i = (long)blockIdx.x * blockDim.x + threadIdx.x;
    if (i < n) out[i] = (_Float16)in[i];
}

// W (K x N, f32) -> Wt (Npad x K, f16), zero-padded rows >= N
__global__ void wtrans_k(const float* __restrict__ w, _Float16* __restrict__ wt,
                         int K, int N, int Npad)
{
    long i = (long)blockIdx.x * blockDim.x + threadIdx.x;
    long total = (long)K * Npad;
    if (i >= total) return;
    int k = (int)(i % K);
    int n = (int)(i / K);
    float v = (n < N) ? w[(long)k * N + n] : 0.f;
    wt[(long)n * K + k] = (_Float16)v;
}

// mvh (B,S,AH) f16 -> mvt (B,H,D,S) f16
__global__ void mvt_k(const _Float16* __restrict__ mvh, _Float16* __restrict__ mvt,
                      long total)
{
    long i = (long)blockIdx.x * blockDim.x + threadIdx.x;
    if (i >= total) return;
    int s = (int)(i % SS);
    long t = i / SS;
    int d = (int)(t % DD); t /= DD;
    int h = (int)(t % NH);
    long b = t / NH;
    mvt[i] = mvh[(b * SS + s) * AH + h * DD + d];
}

// cah = mkch * mqh (conv_attn), elementwise f16
__global__ void mulh_k(const _Float16* __restrict__ a, const _Float16* __restrict__ b,
                       _Float16* __restrict__ o, long n)
{
    long i = (long)blockIdx.x * blockDim.x + threadIdx.x;
    if (i < n) o[i] = (_Float16)((float)a[i] * (float)b[i]);
}

// depthwise conv over sequence dim -> f16
__global__ void depthwise_k(const float* __restrict__ Kin,
                            const float* __restrict__ w,
                            _Float16* __restrict__ dw, long total)
{
    long idx = (long)blockIdx.x * blockDim.x + threadIdx.x;
    if (idx >= total) return;
    int c = (int)(idx % HID);
    long bs = idx / HID;
    int s = (int)(bs % SS);
    long b = bs / SS;
    float acc = 0.f;
    #pragma unroll
    for (int k = 0; k < KS; ++k) {
        int sp = s + k - PADK;
        if (sp >= 0 && sp < SS)
            acc += Kin[(b * SS + sp) * HID + c] * w[c * KS + k];
    }
    dw[idx] = (_Float16)acc;
}

// softmax over the 9-wide kernel axis, in place on ck[(b*S+s)*64 + h*9 + k]
__global__ void cksoftmax_k(float* __restrict__ ck, long rows)
{
    long rid = (long)blockIdx.x * blockDim.x + threadIdx.x;
    if (rid >= rows) return;
    long base = (rid / NH) * 64 + (rid % NH) * KS;
    float v[KS], mx = -1e30f;
    #pragma unroll
    for (int k = 0; k < KS; ++k) { v[k] = ck[base + k]; mx = fmaxf(mx, v[k]); }
    float sm = 0.f;
    #pragma unroll
    for (int k = 0; k < KS; ++k) { v[k] = __expf(v[k] - mx); sm += v[k]; }
    float inv = 1.f / sm;
    #pragma unroll
    for (int k = 0; k < KS; ++k) ck[base + k] = v[k] * inv;
}

// conv_out[b,s,h,d] -> out[b][s][AH + h*64 + d]
__global__ void convout_k(const float* __restrict__ co,
                          const float* __restrict__ ck,
                          float* __restrict__ out, long total)
{
    long idx = (long)blockIdx.x * blockDim.x + threadIdx.x;
    if (idx >= total) return;
    int d = (int)(idx & (DD - 1));
    long t = idx >> 6;
    int h = (int)(t % NH); t /= NH;
    int s = (int)(t % SS);
    long b = t / SS;
    long ckb = (b * SS + s) * 64 + h * KS;
    float acc = 0.f;
    #pragma unroll
    for (int k = 0; k < KS; ++k) {
        int sp = s + k - PADK;
        if (sp >= 0 && sp < SS)
            acc += co[(b * SS + sp) * AH + h * DD + d] * ck[ckb + k];
    }
    out[(b * SS + s) * (2 * AH) + AH + h * DD + d] = acc;
}

// ---- block reduction helpers (256 threads) ----
__device__ __forceinline__ float blk_sum(float v, float* red) {
    int t = threadIdx.x;
    red[t] = v; __syncthreads();
    for (int off = 128; off > 0; off >>= 1) {
        if (t < off) red[t] += red[t + off];
        __syncthreads();
    }
    float r = red[0]; __syncthreads();
    return r;
}
__device__ __forceinline__ float blk_max(float v, float* red) {
    int t = threadIdx.x;
    red[t] = v; __syncthreads();
    for (int off = 128; off > 0; off >>= 1) {
        if (t < off) red[t] = fmaxf(red[t], red[t + off]);
        __syncthreads();
    }
    float r = red[0]; __syncthreads();
    return r;
}

// td_sm[b][j] = softmax_j( mask ? td/||td|| : -1e4 ) ; one block per batch
__global__ __launch_bounds__(256) void tdsm_k(const float* __restrict__ td,
                                              const int* __restrict__ mask,
                                              float* __restrict__ tdsm)
{
    __shared__ float red[256];
    int b = blockIdx.x, t = threadIdx.x;
    float v[4], mkd[4];
    float ss = 0.f;
    #pragma unroll
    for (int e = 0; e < 4; ++e) {
        v[e] = td[b * SS + t * 4 + e];
        ss += v[e] * v[e];
    }
    float den = fmaxf(sqrtf(blk_sum(ss, red)), 1e-12f);
    float mx = -1e30f;
    #pragma unroll
    for (int e = 0; e < 4; ++e) {
        int j = t * 4 + e;
        mkd[e] = mask[b * SS + j] ? (v[e] / den) : -1e4f;
        mx = fmaxf(mx, mkd[e]);
    }
    float rmax = blk_max(mx, red);
    float ls = 0.f;
    #pragma unroll
    for (int e = 0; e < 4; ++e) { mkd[e] = __expf(mkd[e] - rmax); ls += mkd[e]; }
    float inv = 1.f / blk_sum(ls, red);
    #pragma unroll
    for (int e = 0; e < 4; ++e) tdsm[b * SS + t * 4 + e] = mkd[e] * inv;
}

// =====================================================================
// Fused dist_func + rescale + second softmax. One block (256 thr) per
// (b,h,i) score row (S=1024, 4 elems/thread). Writes f16 probs IN PLACE
// over the f32 score row (all reads are in registers before any write;
// one block owns one row). f16 row stride seen by ctx GEMM = 2*S halves.
// =====================================================================
__global__ __launch_bounds__(256) void rowproc_k(float* __restrict__ sc,
                                                 const int* __restrict__ mask,
                                                 const float* __restrict__ tdsm,
                                                 const float* __restrict__ gammas)
{
    __shared__ float red[256];
    long row = blockIdx.x;
    int i = (int)(row % SS);
    long t2 = row / SS;
    int h = (int)(t2 % NH);
    int b = (int)(t2 / NH);
    float* rp = sc + row * SS;
    const int t = threadIdx.x;

    float s[4], p[4];
    int m[4];
    float mx = -1e30f;
    #pragma unroll
    for (int e = 0; e < 4; ++e) {
        int j = t * 4 + e;
        s[e] = rp[j];
        m[e] = mask[b * SS + j];
        float msc = m[e] ? s[e] : -1e8f;
        p[e] = msc;
        mx = fmaxf(mx, msc);
    }
    float rmax = blk_max(mx, red);
    float ls = 0.f;
    #pragma unroll
    for (int e = 0; e < 4; ++e) { p[e] = __expf(p[e] - rmax); ls += p[e]; }
    float inv = 1.f / blk_sum(ls, red);
    float tsum = 0.f;
    float pre[4];
    #pragma unroll
    for (int e = 0; e < 4; ++e) {
        p[e] = p[e] * inv * (m[e] ? 1.f : 0.f);
        tsum += p[e];
        pre[e] = tsum;                       // local inclusive prefix
    }
    // block inclusive scan over per-thread sums (Hillis-Steele)
    red[t] = tsum; __syncthreads();
    for (int off = 1; off < 256; off <<= 1) {
        float add = (t >= off) ? red[t - off] : 0.f;
        __syncthreads();
        red[t] += add;
        __syncthreads();
    }
    float excl = red[t] - tsum;
    float disttotal = red[255];
    __syncthreads();

    float gamma = -log1pf(__expf(gammas[h]));  // -softplus
    float s2[4];
    mx = -1e30f;
    #pragma unroll
    for (int e = 0; e < 4; ++e) {
        int j = t * 4 + e;
        float distcum = excl + pre[e];
        float pos = fabsf((float)(j - i));
        float ds = sqrtf(fmaxf((disttotal - distcum) * pos, 0.f));
        float te = __expf(ds * gamma);
        te = fminf(fmaxf(te, 1e-5f), 1e5f);
        float eff = te - tdsm[b * SS + j] * ((j < i) ? 1.f : 0.f);
        s2[e] = m[e] ? s[e] * eff : -1e8f;
        mx = fmaxf(mx, s2[e]);
    }
    rmax = blk_max(mx, red);
    ls = 0.f;
    #pragma unroll
    for (int e = 0; e < 4; ++e) { s2[e] = __expf(s2[e] - rmax); ls += s2[e]; }
    inv = 1.f / blk_sum(ls, red);
    _Float16* hp = (_Float16*)rp;            // f16 probs in place
    #pragma unroll
    for (int e = 0; e < 4; ++e) hp[t * 4 + e] = (_Float16)(s2[e] * inv);
}

// =====================================================================
extern "C" void kernel_launch(void* const* d_in, const int* in_sizes, int n_in,
                              void* d_out, int out_size, void* d_ws, size_t ws_size,
                              hipStream_t stream)
{
    const float* Q     = (const float*)d_in[0];
    const float* Kin   = (const float*)d_in[1];
    const float* V     = (const float*)d_in[2];
    const float* td    = (const float*)d_in[3];
    const int*   mask  = (const int*)d_in[4];
    const float* Wq    = (const float*)d_in[5];
    const float* Wk    = (const float*)d_in[6];
    const float* Wv    = (const float*)d_in[7];
    const float* dw_w  = (const float*)d_in[8];
    const float* pw_w  = (const float*)d_in[9];
    const float* sep_b = (const float*)d_in[10];
    const float* ck_W  = (const float*)d_in[11];
    const float* ck_b  = (const float*)d_in[12];
    const float* co_W  = (const float*)d_in[13];
    const float* co_b  = (const float*)d_in[14];
    const float* gam   = (const float*)d_in[15];
    float* out = (float*)d_out;

    const long M = (long)BB * SS;   // 4096

    // ---- workspace carve-up (f32 region first for alignment) ----
    float* wf = (float*)d_ws;
    float* co   = wf; wf += M * AH;                 // f32
    float* ckr  = wf; wf += M * 64;                 // f32, padded N=64
    float* tdsm = wf; wf += BB * SS;                // f32
    float* scr  = wf; wf += (long)BB * NH * SS * SS; // f32 scores -> f16 probs
    _Float16* hh = (_Float16*)wf;
    _Float16* Qh   = hh; hh += M * HID;
    _Float16* Kh   = hh; hh += M * HID;
    _Float16* Vh   = hh; hh += M * HID;
    _Float16* dwh  = hh; hh += M * HID;
    _Float16* WqT  = hh; hh += (long)AH * HID;
    _Float16* WkT  = hh; hh += (long)AH * HID;
    _Float16* WvT  = hh; hh += (long)AH * HID;
    _Float16* coWT = hh; hh += (long)AH * HID;
    _Float16* pwh  = hh; hh += (long)AH * HID;
    _Float16* ckWT = hh; hh += (long)64 * AH;
    _Float16* mqh  = hh; hh += M * AH;
    _Float16* mkh  = hh; hh += M * AH;
    _Float16* mvh  = hh; hh += M * AH;
    _Float16* mkch = hh; hh += M * AH;
    _Float16* cah  = hh; hh += M * AH;
    _Float16* mvt  = hh; hh += M * AH;

    dim3 blk(128);
    dim3 gP(AH / 64, (int)(M / 64), 1);     // (6, 64)
    const long nQKV = M * HID;

    // ---- f32 -> f16 conversions and weight transposes ----
    cvt_h_k<<<(int)((nQKV + 255) / 256), 256, 0, stream>>>(Q, Qh, nQKV);
    cvt_h_k<<<(int)((nQKV + 255) / 256), 256, 0, stream>>>(Kin, Kh, nQKV);
    cvt_h_k<<<(int)((nQKV + 255) / 256), 256, 0, stream>>>(V, Vh, nQKV);
    {
        long nw = (long)HID * AH;
        int g = (int)((nw + 255) / 256);
        wtrans_k<<<g, 256, 0, stream>>>(Wq, WqT, HID, AH, AH);
        wtrans_k<<<g, 256, 0, stream>>>(Wk, WkT, HID, AH, AH);
        wtrans_k<<<g, 256, 0, stream>>>(Wv, WvT, HID, AH, AH);
        wtrans_k<<<g, 256, 0, stream>>>(co_W, coWT, HID, AH, AH);
        cvt_h_k<<<g, 256, 0, stream>>>(pw_w, pwh, nw);
        long nck = (long)AH * 64;
        wtrans_k<<<(int)((nck + 255) / 256), 256, 0, stream>>>(ck_W, ckWT, AH, NH * KS, 64);
    }
    depthwise_k<<<(int)((nQKV + 255) / 256), 256, 0, stream>>>(Kin, dw_w, dwh, nQKV);

    // ---- projections: 4096 x 384 x 768 (f16 outputs) ----
    gemm_wmma_f16<<<gP, blk, 0, stream>>>(Qh, HID, WqT, HID, nullptr, mqh, AH,
        nullptr, AH, HID, 1.f, 1, 0, 0, 0, 0, 0, 0);
    gemm_wmma_f16<<<gP, blk, 0, stream>>>(Kh, HID, WkT, HID, nullptr, mkh, AH,
        nullptr, AH, HID, 1.f, 1, 0, 0, 0, 0, 0, 0);
    gemm_wmma_f16<<<gP, blk, 0, stream>>>(Vh, HID, WvT, HID, nullptr, mvh, AH,
        nullptr, AH, HID, 1.f, 1, 0, 0, 0, 0, 0, 0);
    gemm_wmma_f16<<<gP, blk, 0, stream>>>(Vh, HID, coWT, HID, co, nullptr, AH,
        co_b, AH, HID, 1.f, 1, 0, 0, 0, 0, 0, 0);
    // pointwise conv: dwh @ pw_w^T + sep_b -> mkch (f16)
    gemm_wmma_f16<<<gP, blk, 0, stream>>>(dwh, HID, pwh, HID, nullptr, mkch, AH,
        sep_b, AH, HID, 1.f, 1, 0, 0, 0, 0, 0, 0);

    // ---- dynamic span kernel ----
    mulh_k<<<(int)((M * AH + 255) / 256), 256, 0, stream>>>(mkch, mqh, cah, M * AH);
    {
        dim3 g(1, (int)(M / 64), 1);   // N padded to 64
        gemm_wmma_f16<<<g, blk, 0, stream>>>(cah, AH, ckWT, AH, ckr, nullptr, 64,
            ck_b, NH * KS, AH, 1.f, 1, 0, 0, 0, 0, 0, 0);
    }
    cksoftmax_k<<<(int)((M * NH + 255) / 256), 256, 0, stream>>>(ckr, M * NH);
    convout_k<<<(int)((M * NH * DD + 255) / 256), 256, 0, stream>>>(co, ckr, out,
                                                                    M * NH * DD);

    // ---- attention ----
    mvt_k<<<(int)((M * AH + 255) / 256), 256, 0, stream>>>(mvh, mvt, M * AH);
    tdsm_k<<<BB, 256, 0, stream>>>(td, mask, tdsm);
    {   // scores = q @ k^T / 8, batched over (b,h)
        dim3 g(SS / 64, SS / 64, BB * NH);
        gemm_wmma_f16<<<g, blk, 0, stream>>>(
            mqh, AH, mkh, AH, scr, nullptr, SS, nullptr, SS, DD, 0.125f,
            NH, (long)SS * AH, (long)DD,
                (long)SS * AH, (long)DD,
                (long)NH * SS * SS, (long)SS * SS);
    }
    rowproc_k<<<(int)((long)BB * NH * SS), 256, 0, stream>>>(scr, mask, tdsm, gam);
    {   // ctx = probs(f16, stride 2S) @ v -> out[:, :, 0:384]
        dim3 g(1, SS / 64, BB * NH);
        gemm_wmma_f16<<<g, blk, 0, stream>>>(
            (const _Float16*)scr, 2 * SS, mvt, SS, out, nullptr, 2 * AH,
            nullptr, 2 * AH, SS, 1.f,
            NH, (long)NH * 2 * SS * SS, (long)2 * SS * SS,
                (long)NH * DD * SS, (long)DD * SS,
                (long)SS * (2 * AH), (long)DD);
    }
    (void)in_sizes; (void)n_in; (void)out_size; (void)ws_size;
}
